// SparseMoe_8624294331061
// MI455X (gfx1250) — matrix operand order repaired and checked by reading, not compiled, not run on hardware
//
#include <hip/hip_runtime.h>

// ---------------------------------------------------------------------------
// MoE forward for MI455X (gfx1250, wave32, WMMA + TDM).
// gate -> one-time bf16 convert of x + transposed bf16 weights -> per expert:
//   ffn1: h = silu(x@W1) * (x@W3)   (fused dual-GEMM, bf16 WMMA, f32 accum)
//   ffn2: y += w[:,e] * (h@W2)
// Tiles are DMA'd into double-buffered LDS by the Tensor Data Mover
// (tensor_load_to_lds / TENSORcnt); steady-state loop = ds_load_b128 + wmma
// at a 1.0 load:wmma ratio (64x32 resp. 64x64 wave tiles).
// ---------------------------------------------------------------------------

#define N_TOK 8192   // B*T
#define C_DIM 2048
#define F_DIM 5632
#define N_EXP 8
#define BK    32

typedef __bf16 bf16;
typedef __attribute__((ext_vector_type(16))) bf16  v16bf;
typedef __attribute__((ext_vector_type(8)))  bf16  v8bf;
typedef __attribute__((ext_vector_type(8)))  float v8f;
typedef __attribute__((ext_vector_type(4)))  float v4f;
typedef __attribute__((ext_vector_type(4)))  unsigned int v4u;
typedef __attribute__((ext_vector_type(4)))  int  v4i;
typedef __attribute__((ext_vector_type(8)))  int  v8i;

#if __has_builtin(__builtin_amdgcn_tensor_load_to_lds) && \
    __has_builtin(__builtin_amdgcn_s_wait_tensorcnt)
#define HAVE_TDM 1
#else
#define HAVE_TDM 0
#endif

__device__ __forceinline__ bf16 f2bf(float f) {
  unsigned u = __builtin_bit_cast(unsigned, f);
  unsigned r = (u + 0x7FFFu + ((u >> 16) & 1u)) >> 16;   // round-nearest-even
  unsigned short s = (unsigned short)r;
  return __builtin_bit_cast(bf16, s);
}

#if HAVE_TDM
// 2-D TDM load: tile_d1 rows x tile_d0 bf16 elems (contiguous), global row
// stride = stride_elems, packed row-major into LDS at lds_off. D# per ISA ch.8.
__device__ __forceinline__ void tdm_load_2d(const void* gptr, unsigned lds_off,
                                            unsigned stride_elems,
                                            unsigned tile_d0, unsigned tile_d1) {
  unsigned long long ga = (unsigned long long)gptr;
  unsigned td0 = stride_elems;     // tensor width (row length, elem units)
  unsigned td1 = 0x100000u;        // generous row count: OOB clip never trips
  v4u g0 = { 1u,                                    // count=1 valid descriptor
             lds_off,                               // LDS byte address
             (unsigned)ga,                          // global_addr[31:0]
             (unsigned)((ga >> 32) & 0x01FFFFFFu)   // global_addr[56:32]
                 | 0x80000000u };                   // type=2 ("image")
  v8i g1 = { (int)(1u << 16),                       // data_size=1 -> 2 bytes
             (int)((td0 & 0xFFFFu) << 16),          // tensor_dim0[15:0]
             (int)((td0 >> 16) | ((td1 & 0xFFFFu) << 16)),
             (int)((td1 >> 16) | (tile_d0 << 16)),  // tile_dim0
             (int)(tile_d1 & 0xFFFFu),              // tile_dim1 (tile_dim2=0)
             (int)stride_elems,                     // tensor_dim0_stride[31:0]
             0, 0 };
  v4i z4 = { 0, 0, 0, 0 };
#if defined(__clang_major__) && (__clang_major__ >= 23)
  v8i z8 = { 0, 0, 0, 0, 0, 0, 0, 0 };
  __builtin_amdgcn_tensor_load_to_lds(g0, g1, z4, z4, z8, 0);
#else
  __builtin_amdgcn_tensor_load_to_lds(g0, g1, z4, z4, 0);
#endif
}
#endif  // HAVE_TDM

// ---------------------------------------------------------------------------
// Gating: one wave32 per token. logits = x_row @ Wg, softmax, top-2, renorm.
// ---------------------------------------------------------------------------
__global__ __launch_bounds__(256)
void gate_kernel(const float* __restrict__ x, const float* __restrict__ Wg,
                 float* __restrict__ wout) {
  int lane = threadIdx.x & 31;
  int tok  = (blockIdx.x * blockDim.x + threadIdx.x) >> 5;
  if (tok >= N_TOK) return;
  const float* xr = x + (size_t)tok * C_DIM;

  float acc[N_EXP];
#pragma unroll
  for (int e = 0; e < N_EXP; ++e) acc[e] = 0.f;
  for (int c = lane; c < C_DIM; c += 32) {
    float xv = xr[c];
    const float* wg = Wg + (size_t)c * N_EXP;
#pragma unroll
    for (int e = 0; e < N_EXP; ++e) acc[e] += xv * wg[e];
  }
#pragma unroll
  for (int off = 16; off >= 1; off >>= 1) {
#pragma unroll
    for (int e = 0; e < N_EXP; ++e) acc[e] += __shfl_xor(acc[e], off, 32);
  }
  if (lane == 0) {
    float mx = acc[0];
#pragma unroll
    for (int e = 1; e < N_EXP; ++e) mx = fmaxf(mx, acc[e]);
    float p[N_EXP], sum = 0.f;
#pragma unroll
    for (int e = 0; e < N_EXP; ++e) { p[e] = __expf(acc[e] - mx); sum += p[e]; }
    float inv = 1.f / sum;
#pragma unroll
    for (int e = 0; e < N_EXP; ++e) p[e] *= inv;
    int i0 = 0;
#pragma unroll
    for (int e = 1; e < N_EXP; ++e) if (p[e] > p[i0]) i0 = e;
    int i1 = (i0 == 0) ? 1 : 0;
#pragma unroll
    for (int e = 0; e < N_EXP; ++e) if (e != i0 && p[e] > p[i1]) i1 = e;
    float ps = p[i0] + p[i1];
    float o[N_EXP];
#pragma unroll
    for (int e = 0; e < N_EXP; ++e) o[e] = 0.f;
    o[i0] = p[i0] / ps;
    o[i1] = p[i1] / ps;
    float* wrow = wout + (size_t)tok * N_EXP;
#pragma unroll
    for (int e = 0; e < N_EXP; ++e) wrow[e] = o[e];
  }
}

// ---------------------------------------------------------------------------
// x fp32 -> bf16 (row-major, A operand layout)
// ---------------------------------------------------------------------------
__global__ __launch_bounds__(256)
void cvt_kernel(const float* __restrict__ x, bf16* __restrict__ xb, int n) {
  int i = (blockIdx.x * blockDim.x + threadIdx.x) * 4;
  if (i + 3 < n) {
    v4f v = *(const v4f*)(x + i);
#pragma unroll
    for (int q = 0; q < 4; ++q) xb[i + q] = f2bf(v[q]);
  }
}

// ---------------------------------------------------------------------------
// One-time weight transform: src fp32 [R][S] (one matrix per blockIdx.z)
// -> dst bf16 [S][R]  (B operand layout [n][k], k contiguous).
// ---------------------------------------------------------------------------
__global__ __launch_bounds__(256)
void transp_cvt_kernel(const float* __restrict__ src, bf16* __restrict__ dst,
                       int R, int S) {
  __shared__ float tile[32][33];
  const float* s = src + (size_t)blockIdx.z * R * S;
  bf16*        d = dst + (size_t)blockIdx.z * R * S;
  int r0 = blockIdx.y * 32, c0 = blockIdx.x * 32;
  int tx = threadIdx.x & 31, ty = threadIdx.x >> 5;   // ty: 0..7
#pragma unroll
  for (int i = 0; i < 32; i += 8)
    tile[ty + i][tx] = s[(size_t)(r0 + ty + i) * S + c0 + tx];
  __syncthreads();
#pragma unroll
  for (int i = 0; i < 32; i += 8)
    d[(size_t)(c0 + ty + i) * R + r0 + tx] = f2bf(tile[tx][ty + i]);
}

// ---------------------------------------------------------------------------
// Fragment loads (ISA 7.12.2 layouts).
// ---------------------------------------------------------------------------
__device__ __forceinline__ v16bf load_afrag(const bf16* rowp, int lane) {
  int klo = (lane < 16) ? 0 : 8;
  v8bf lo = *(const v8bf*)(rowp + klo);
  v8bf hi = *(const v8bf*)(rowp + klo + 16);
  v16bf a;
#pragma unroll
  for (int q = 0; q < 8; ++q) { a[q] = lo[q]; a[8 + q] = hi[q]; }
  return a;
}
__device__ __forceinline__ v16bf load_bfrag(const bf16* colp, int lane) {
  int kb = (lane < 16) ? 0 : 8;
  v8bf lo = *(const v8bf*)(colp + kb);
  v8bf hi = *(const v8bf*)(colp + kb + 16);
  v16bf b;
#pragma unroll
  for (int q = 0; q < 8; ++q) { b[q] = lo[q]; b[8 + q] = hi[q]; }
  return b;
}

__device__ __forceinline__ unsigned lds_off_of(const void* p) {
  return (unsigned)(unsigned long long)p;   // low 32 bits = LDS byte offset
}

// ---------------------------------------------------------------------------
// ffn1: h = silu(x@W1t^T) * (x@W3t^T).
// Block tile 128(M) x 128(N), 8 waves (2x4); wave tile 64x32 per matmul.
// Per K-step per wave: 16 ds_load_b128, 16 wmma.
// ---------------------------------------------------------------------------
__global__ __launch_bounds__(256)
void ffn1_kernel(const bf16* __restrict__ xb,    // [N_TOK][C]
                 const bf16* __restrict__ W1t,   // [F][C]
                 const bf16* __restrict__ W3t,   // [F][C]
                 bf16* __restrict__ h) {         // [N_TOK][F]
  __shared__ __align__(16) bf16 ldsA [2][128][BK];   // 16 KB
  __shared__ __align__(16) bf16 ldsB1[2][128][BK];   // 16 KB
  __shared__ __align__(16) bf16 ldsB3[2][128][BK];   // 16 KB

  const int tid  = threadIdx.x;
  const int lane = tid & 31;
  const int wid  = tid >> 5;
  const int wr   = wid >> 2;          // 0..1 -> 64-row slab
  const int wc   = wid & 3;           // 0..3 -> 32-col slab
  const int m0   = blockIdx.x * 128;
  const int n0   = blockIdx.y * 128;
  const int KS   = C_DIM / BK;

  v8f zero = {};
  v8f acc1[4][2], acc3[4][2];
#pragma unroll
  for (int i = 0; i < 4; ++i)
#pragma unroll
    for (int j = 0; j < 2; ++j) { acc1[i][j] = zero; acc3[i][j] = zero; }

#if HAVE_TDM
  if (wid == 0) {   // prologue: DMA step-0 tiles into buffer 0
    tdm_load_2d(xb  + (size_t)m0 * C_DIM, lds_off_of(&ldsA [0][0][0]), C_DIM, BK, 128);
    tdm_load_2d(W1t + (size_t)n0 * C_DIM, lds_off_of(&ldsB1[0][0][0]), C_DIM, BK, 128);
    tdm_load_2d(W3t + (size_t)n0 * C_DIM, lds_off_of(&ldsB3[0][0][0]), C_DIM, BK, 128);
  }
#endif

  for (int it = 0; it < KS; ++it) {
    const int buf = it & 1;
#if HAVE_TDM
    if (wid == 0) {
      if (it + 1 < KS) {     // prefetch next tiles into the other buffer
        const int nb = (it + 1) & 1;
        const int k1 = (it + 1) * BK;
        tdm_load_2d(xb  + (size_t)m0 * C_DIM + k1, lds_off_of(&ldsA [nb][0][0]), C_DIM, BK, 128);
        tdm_load_2d(W1t + (size_t)n0 * C_DIM + k1, lds_off_of(&ldsB1[nb][0][0]), C_DIM, BK, 128);
        tdm_load_2d(W3t + (size_t)n0 * C_DIM + k1, lds_off_of(&ldsB3[nb][0][0]), C_DIM, BK, 128);
        __builtin_amdgcn_s_wait_tensorcnt(3);   // current group done
      } else {
        __builtin_amdgcn_s_wait_tensorcnt(0);
      }
    }
#else
    { // manual staging fallback: pure 16B copies
      const int k0 = it * BK;
      int r = tid >> 1, k = (tid & 1) * 16;
      *(v8bf*)&ldsA [buf][r][k]     = *(const v8bf*)(xb  + (size_t)(m0 + r) * C_DIM + k0 + k);
      *(v8bf*)&ldsA [buf][r][k + 8] = *(const v8bf*)(xb  + (size_t)(m0 + r) * C_DIM + k0 + k + 8);
      *(v8bf*)&ldsB1[buf][r][k]     = *(const v8bf*)(W1t + (size_t)(n0 + r) * C_DIM + k0 + k);
      *(v8bf*)&ldsB1[buf][r][k + 8] = *(const v8bf*)(W1t + (size_t)(n0 + r) * C_DIM + k0 + k + 8);
      *(v8bf*)&ldsB3[buf][r][k]     = *(const v8bf*)(W3t + (size_t)(n0 + r) * C_DIM + k0 + k);
      *(v8bf*)&ldsB3[buf][r][k + 8] = *(const v8bf*)(W3t + (size_t)(n0 + r) * C_DIM + k0 + k + 8);
    }
#endif
    __syncthreads();

    const int arow = lane & 15;
    v16bf afrag[4];
#pragma unroll
    for (int mi = 0; mi < 4; ++mi)
      afrag[mi] = load_afrag(&ldsA[buf][wr * 64 + mi * 16 + arow][0], lane);

    const int bcol = lane & 15;
    v16bf b1f[2], b3f[2];
#pragma unroll
    for (int ni = 0; ni < 2; ++ni) {
      b1f[ni] = load_bfrag(&ldsB1[buf][wc * 32 + ni * 16 + bcol][0], lane);
      b3f[ni] = load_bfrag(&ldsB3[buf][wc * 32 + ni * 16 + bcol][0], lane);
    }

#pragma unroll
    for (int mi = 0; mi < 4; ++mi)
#pragma unroll
      for (int ni = 0; ni < 2; ++ni) {
        acc1[mi][ni] = __builtin_amdgcn_wmma_f32_16x16x32_bf16(
            false, afrag[mi], false, b1f[ni], (short)0, acc1[mi][ni], false, false);
        acc3[mi][ni] = __builtin_amdgcn_wmma_f32_16x16x32_bf16(
            false, afrag[mi], false, b3f[ni], (short)0, acc3[mi][ni], false, false);
      }
    __syncthreads();   // all reads of buf done before TDM overwrites it
  }

  // epilogue: h = silu(acc1) * acc3; C/D elem i -> row i+8*laneHi, col lane&15
  const int ccol = lane & 15;
  const int chi  = (lane >> 4) & 1;
#pragma unroll
  for (int mi = 0; mi < 4; ++mi)
#pragma unroll
    for (int ni = 0; ni < 2; ++ni)
#pragma unroll
      for (int i = 0; i < 8; ++i) {
        int m = m0 + wr * 64 + mi * 16 + i + chi * 8;
        int n = n0 + wc * 32 + ni * 16 + ccol;
        float g = acc1[mi][ni][i];
        float s = g * __builtin_amdgcn_rcpf(1.f + __expf(-g));   // fast silu
        h[(size_t)m * F_DIM + n] = f2bf(s * acc3[mi][ni][i]);
      }
}

// ---------------------------------------------------------------------------
// ffn2: y += w[:,e] * (h @ W2t^T).
// Block tile 128(M) x 256(N), 8 waves (2x4); wave tile 64x64.
// Per K-step per wave: 16 ds_load_b128, 16 wmma.
// ---------------------------------------------------------------------------
__global__ __launch_bounds__(256)
void ffn2_kernel(const bf16* __restrict__ h,     // [N_TOK][F]
                 const bf16* __restrict__ W2t,   // [C][F]
                 const float* __restrict__ wgate, int expert,
                 float* __restrict__ y) {
  __shared__ __align__(16) bf16 ldsA[2][128][BK];   // 16 KB
  __shared__ __align__(16) bf16 ldsB[2][256][BK];   // 32 KB

  const int tid  = threadIdx.x;
  const int lane = tid & 31;
  const int wid  = tid >> 5;
  const int wr   = wid >> 2;          // 0..1 -> 64-row slab
  const int wc   = wid & 3;           // 0..3 -> 64-col slab
  const int m0   = blockIdx.x * 128;
  const int n0   = blockIdx.y * 256;
  const int KS   = F_DIM / BK;

  v8f zero = {};
  v8f acc[4][4];
#pragma unroll
  for (int i = 0; i < 4; ++i)
#pragma unroll
    for (int j = 0; j < 4; ++j) acc[i][j] = zero;

#if HAVE_TDM
  if (wid == 0) {
    tdm_load_2d(h   + (size_t)m0 * F_DIM, lds_off_of(&ldsA[0][0][0]), F_DIM, BK, 128);
    tdm_load_2d(W2t + (size_t)n0 * F_DIM, lds_off_of(&ldsB[0][0][0]), F_DIM, BK, 256);
  }
#endif

  for (int it = 0; it < KS; ++it) {
    const int buf = it & 1;
#if HAVE_TDM
    if (wid == 0) {
      if (it + 1 < KS) {
        const int nb = (it + 1) & 1;
        const int k1 = (it + 1) * BK;
        tdm_load_2d(h   + (size_t)m0 * F_DIM + k1, lds_off_of(&ldsA[nb][0][0]), F_DIM, BK, 128);
        tdm_load_2d(W2t + (size_t)n0 * F_DIM + k1, lds_off_of(&ldsB[nb][0][0]), F_DIM, BK, 256);
        __builtin_amdgcn_s_wait_tensorcnt(2);
      } else {
        __builtin_amdgcn_s_wait_tensorcnt(0);
      }
    }
#else
    {
      const int k0 = it * BK;
      int r = tid >> 1, k = (tid & 1) * 16;
      *(v8bf*)&ldsA[buf][r][k]     = *(const v8bf*)(h + (size_t)(m0 + r) * F_DIM + k0 + k);
      *(v8bf*)&ldsA[buf][r][k + 8] = *(const v8bf*)(h + (size_t)(m0 + r) * F_DIM + k0 + k + 8);
      const bf16* wrow = W2t + (size_t)(n0 + tid) * F_DIM + k0;
#pragma unroll
      for (int q = 0; q < 4; ++q)
        *(v8bf*)&ldsB[buf][tid][q * 8] = *(const v8bf*)(wrow + q * 8);
    }
#endif
    __syncthreads();

    const int arow = lane & 15;
    v16bf afrag[4];
#pragma unroll
    for (int mi = 0; mi < 4; ++mi)
      afrag[mi] = load_afrag(&ldsA[buf][wr * 64 + mi * 16 + arow][0], lane);

    const int bcol = lane & 15;
    v16bf bfB[4];
#pragma unroll
    for (int ni = 0; ni < 4; ++ni)
      bfB[ni] = load_bfrag(&ldsB[buf][wc * 64 + ni * 16 + bcol][0], lane);

#pragma unroll
    for (int mi = 0; mi < 4; ++mi)
#pragma unroll
      for (int ni = 0; ni < 4; ++ni)
        acc[mi][ni] = __builtin_amdgcn_wmma_f32_16x16x32_bf16(
            false, afrag[mi], false, bfB[ni], (short)0, acc[mi][ni], false, false);
    __syncthreads();
  }

  const int ccol = lane & 15;
  const int chi  = (lane >> 4) & 1;
#pragma unroll
  for (int mi = 0; mi < 4; ++mi)
#pragma unroll
    for (int ni = 0; ni < 4; ++ni)
#pragma unroll
      for (int i = 0; i < 8; ++i) {
        int m = m0 + wr * 64 + mi * 16 + i + chi * 8;
        int n = n0 + wc * 64 + ni * 16 + ccol;
        float wg = wgate[(size_t)m * N_EXP + expert];
        float* yp = y + (size_t)m * C_DIM + n;
        *yp += wg * acc[mi][ni][i];   // experts sequential -> safe RMW
      }
}

// ---------------------------------------------------------------------------
extern "C" void kernel_launch(void* const* d_in, const int* in_sizes, int n_in,
                              void* d_out, int out_size, void* d_ws, size_t ws_size,
                              hipStream_t stream) {
  const float* x  = (const float*)d_in[0];   // [B,T,C]
  const float* Wg = (const float*)d_in[1];   // [C,E]
  const float* W1 = (const float*)d_in[2];   // [E,C,F]
  const float* W3 = (const float*)d_in[3];   // [E,C,F]
  const float* W2 = (const float*)d_in[4];   // [E,F,C]
  float* y = (float*)d_out;                  // [B,T,C]

  // workspace layout (~678 MB): gate w | x bf16 | h bf16 | W1t | W3t | W2t
  char* ws = (char*)d_ws;
  size_t off = 0;
  float* wgate = (float*)(ws + off);
  off += (size_t)N_TOK * N_EXP * sizeof(float);
  off = (off + 255) & ~(size_t)255;
  bf16* xb = (bf16*)(ws + off);
  off += (size_t)N_TOK * C_DIM * sizeof(bf16);
  off = (off + 255) & ~(size_t)255;
  bf16* hbuf = (bf16*)(ws + off);
  off += (size_t)N_TOK * F_DIM * sizeof(bf16);
  off = (off + 255) & ~(size_t)255;
  bf16* W1t = (bf16*)(ws + off);             // [E][F][C]
  off += (size_t)N_EXP * F_DIM * C_DIM * sizeof(bf16);
  bf16* W3t = (bf16*)(ws + off);             // [E][F][C]
  off += (size_t)N_EXP * F_DIM * C_DIM * sizeof(bf16);
  bf16* W2t = (bf16*)(ws + off);             // [E][C][F]

  hipMemsetAsync(y, 0, (size_t)N_TOK * C_DIM * sizeof(float), stream);

  gate_kernel<<<N_TOK / 8, 256, 0, stream>>>(x, Wg, wgate);
  {
    int n = N_TOK * C_DIM;
    cvt_kernel<<<(n / 4 + 255) / 256, 256, 0, stream>>>(x, xb, n);
  }
  // one-time weight transpose+convert (~48 us of HBM traffic at 23.3 TB/s)
  transp_cvt_kernel<<<dim3(F_DIM / 32, C_DIM / 32, N_EXP), 256, 0, stream>>>(
      W1, W1t, C_DIM, F_DIM);
  transp_cvt_kernel<<<dim3(F_DIM / 32, C_DIM / 32, N_EXP), 256, 0, stream>>>(
      W3, W3t, C_DIM, F_DIM);
  transp_cvt_kernel<<<dim3(C_DIM / 32, F_DIM / 32, N_EXP), 256, 0, stream>>>(
      W2, W2t, F_DIM, C_DIM);

  dim3 g1(N_TOK / 128, F_DIM / 128);   // 64 x 44
  dim3 g2(N_TOK / 128, C_DIM / 256);   // 64 x 8
  for (int e = 0; e < N_EXP; ++e) {
    const bf16* W1te = W1t + (size_t)e * F_DIM * C_DIM;
    const bf16* W3te = W3t + (size_t)e * F_DIM * C_DIM;
    const bf16* W2te = W2t + (size_t)e * C_DIM * F_DIM;
    ffn1_kernel<<<g1, 256, 0, stream>>>(xb, W1te, W3te, hbuf);
    ffn2_kernel<<<g2, 256, 0, stream>>>(hbuf, W2te, wgate, e, y);
  }
}
